// GCNContext_49907519980185
// MI455X (gfx1250) — matrix operand order repaired and checked by reading, not compiled, not run on hardware
//
#include <hip/hip_runtime.h>
#include <stdint.h>

// ---------- CDNA5 WMMA types ----------
typedef __attribute__((ext_vector_type(16))) __bf16 v16bf;
typedef __attribute__((ext_vector_type(8)))  float  v8f;
union Frag16 { uint32_t u[8]; v16bf v; };

// ---------- problem constants (match reference) ----------
constexpr int BGRAPH = 2048;
constexpr int TURNS  = 50;
constexpr int EPERG  = 600;
constexpr int NNODE  = BGRAPH * TURNS;   // 102400
constexpr int INF    = 512;
constexpr int POSF   = 64;
constexpr int K1F    = INF + POSF;       // 576
constexpr float BNEPS = 1e-5f;

__device__ __forceinline__ unsigned short f2bf(float f) {
  union { float f; uint32_t u; } v; v.f = f;
  return (unsigned short)((v.u + 0x7fffu + ((v.u >> 16) & 1u)) >> 16);  // RNE
}

// ---------- CDNA5 async global->LDS helpers ----------
__device__ __forceinline__ void wait_async0() {
#if __has_builtin(__builtin_amdgcn_s_wait_asynccnt)
  __builtin_amdgcn_s_wait_asynccnt(0);
#else
  asm volatile("s_wait_asynccnt 0" ::: "memory");
#endif
}
__device__ __forceinline__ void async_ld_b64(uint32_t lds_addr, uint32_t goff, const void* base) {
  asm volatile("global_load_async_to_lds_b64 %0, %1, %2"
               :: "v"(lds_addr), "v"(goff), "s"(base) : "memory");
}
__device__ __forceinline__ void async_ld_b128(uint32_t lds_addr, uint32_t goff, const void* base) {
  asm volatile("global_load_async_to_lds_b128 %0, %1, %2"
               :: "v"(lds_addr), "v"(goff), "s"(base) : "memory");
}

// ---------- build concat(x, posemb[pos]) as bf16 [N, 576]; one row per block ----------
__global__ void __launch_bounds__(256) build_xc(const float* __restrict__ x,
                                                const int* __restrict__ pos,
                                                const float* __restrict__ posemb,
                                                unsigned short* __restrict__ xc) {
  const int row = blockIdx.x, t = threadIdx.x;
  const float* xr = x + (size_t)row * INF;
  unsigned short* out = xc + (size_t)row * K1F;
  out[t]       = f2bf(xr[t]);
  out[t + 256] = f2bf(xr[t + 256]);
  if (t < POSF) out[INF + t] = f2bf(posemb[pos[row] * POSF + t]);
}

// ---------- pack W [K,512] f32 -> fragment-ready bf16 pairs ----------
// Wq[((ks*2 + hi)*512 + c)*8 + v] = pack(W[2*k2][c], W[2*k2+1][c]),  k2 = ks*16 + (v>>2)*8 + hi*4 + (v&3)
// => each lane's 8-word B fragment is 32 contiguous bytes; a half-wave reads 512 dense bytes.
__global__ void pack_weight(const float* __restrict__ W, uint32_t* __restrict__ Wq, int K) {
  const int total = (K >> 5) * 8192;
  for (int i = blockIdx.x * blockDim.x + threadIdx.x; i < total;
       i += gridDim.x * blockDim.x) {
    const int v  = i & 7;
    const int c  = (i >> 3) & 511;
    const int hi = (i >> 12) & 1;
    const int ks = i >> 13;
    const int k2 = ks * 16 + (v >> 2) * 8 + hi * 4 + (v & 3);
    const uint32_t lo = f2bf(W[(size_t)(2 * k2) * 512 + c]);
    const uint32_t hw = f2bf(W[(size_t)(2 * k2 + 1) * 512 + c]);
    Wq[i] = lo | (hw << 16);
  }
}

// ---------- per-graph dense normalized adjacency Ahat[dst][src] (bf16, 64x64 padded) ----------
__global__ void build_adj(const int* __restrict__ esrc, const int* __restrict__ edst,
                          const float* __restrict__ ew, unsigned short* __restrict__ ahat) {
  __shared__ int   ls[EPERG];
  __shared__ int   ld[EPERG];
  __shared__ float lw[EPERG];
  __shared__ float dinv[TURNS];
  __shared__ float adjs[64][64];
  const int g = blockIdx.x, tid = threadIdx.x;
  const int base = g * EPERG, off = g * TURNS;
  for (int e = tid; e < EPERG; e += blockDim.x) {
    ls[e] = esrc[base + e] - off;
    ld[e] = edst[base + e] - off;
    lw[e] = ew[base + e];
  }
  __syncthreads();
  if (tid < TURNS) {
    float deg = 1.0f;                       // self-loop weight 1
    for (int e = 0; e < EPERG; ++e) deg += (ld[e] == tid) ? lw[e] : 0.0f;
    dinv[tid] = rsqrtf(deg);
  }
  __syncthreads();
  if (tid < 64) {
    for (int s = 0; s < 64; ++s) adjs[tid][s] = 0.0f;
    if (tid < TURNS) {
      const float di = dinv[tid];
      adjs[tid][tid] = di * di;             // self loop
      for (int e = 0; e < EPERG; ++e)
        if (ld[e] == tid) adjs[tid][ls[e]] += dinv[ls[e]] * lw[e] * di;
    }
  }
  __syncthreads();
  unsigned short* out = ahat + (size_t)g * 64 * 64;
  for (int i = tid; i < 64 * 64; i += blockDim.x) out[i] = f2bf(adjs[i >> 6][i & 63]);
}

// ---------- zero the 64 bf16 pad rows after H ----------
__global__ void zero_rows(unsigned short* p) {
  for (int i = blockIdx.x * blockDim.x + threadIdx.x; i < 64 * 512;
       i += gridDim.x * blockDim.x) p[i] = 0;
}

// ---------- main GEMM: C[M x 512] = A(bf16 [M x K]) @ Wq(fragment-packed) ----------
// Block tile 128x128, 8 waves (4 M x 2 N), wave tile 32x64.
// A: async global->LDS DMA, double buffered (72B row stride). B: direct global b128 frags.
constexpr uint32_t ABUF = 128 * 72;          // one A buffer (bytes)
template<bool OUTB16, bool ACC, bool BIAS, bool TANH>
__global__ void __launch_bounds__(256)
gemm_bf16(const unsigned short* __restrict__ A, const uint32_t* __restrict__ Wq,
          int K, unsigned short* outb, float* outf, const float* accin,
          const float* __restrict__ bias) {
  extern __shared__ char smem[];             // 2 * ABUF
  const int tid  = threadIdx.x;
  const int row0 = blockIdx.y * 128;
  const int col0 = blockIdx.x * 128;
  const int wave = tid >> 5, lane = tid & 31;
  const int wm = wave & 3, wn = wave >> 2;
  const int r = lane & 15, hi = lane >> 4;

  // per-thread A-copy tasks: 128 rows x 8 b64 chunks = 1024, 4 per thread
  uint32_t a_goff[4], a_lds[4];
  #pragma unroll
  for (int t2 = 0; t2 < 4; ++t2) {
    const int task = tid + t2 * 256;
    const int ar = task >> 3, ac = task & 7;
    a_goff[t2] = (uint32_t)((row0 + ar) * K) * 2u + (uint32_t)ac * 8u;
    a_lds[t2]  = (uint32_t)(ar * 72 + ac * 8);
  }

  v8f acc[2][4];
  for (int mt = 0; mt < 2; ++mt)
    for (int nt = 0; nt < 4; ++nt)
      for (int e = 0; e < 8; ++e) acc[mt][nt][e] = 0.0f;

  // prefetch tile 0 into buffer 0
  #pragma unroll
  for (int t2 = 0; t2 < 4; ++t2) async_ld_b64(a_lds[t2], a_goff[t2], A);

  const int nks = K >> 5;
  for (int ks = 0; ks < nks; ++ks) {
    const uint32_t cur = (ks & 1) ? ABUF : 0u;
    wait_async0();
    __syncthreads();                         // tile ks landed; buffers safe to swap
    if (ks + 1 < nks) {
      const uint32_t koff = (uint32_t)((ks + 1) << 6);   // bytes along K
      #pragma unroll
      for (int t2 = 0; t2 < 4; ++t2)
        async_ld_b64((cur ^ ABUF) + a_lds[t2], a_goff[t2] + koff, A);
    }

    // B fragments straight from global (L2-resident packed weights)
    Frag16 bfv[4];
    #pragma unroll
    for (int nt = 0; nt < 4; ++nt) {
      const int col = col0 + wn * 64 + nt * 16 + r;
      const uint4* bp = (const uint4*)(Wq + ((size_t)((ks * 2 + hi) * 512 + col)) * 8);
      const uint4 b0 = bp[0], b1 = bp[1];
      bfv[nt].u[0] = b0.x; bfv[nt].u[1] = b0.y; bfv[nt].u[2] = b0.z; bfv[nt].u[3] = b0.w;
      bfv[nt].u[4] = b1.x; bfv[nt].u[5] = b1.y; bfv[nt].u[6] = b1.z; bfv[nt].u[7] = b1.w;
    }
    // A fragments from LDS
    Frag16 af[2];
    #pragma unroll
    for (int mt = 0; mt < 2; ++mt) {
      const char* bp = smem + cur + (wm * 32 + mt * 16 + r) * 72 + hi * 16;
      #pragma unroll
      for (int p = 0; p < 8; ++p)
        af[mt].u[p] = *(const uint32_t*)(bp + p * 4 + ((p >= 4) ? 16 : 0));
    }
    #pragma unroll
    for (int mt = 0; mt < 2; ++mt)
      #pragma unroll
      for (int nt = 0; nt < 4; ++nt)
        acc[mt][nt] = __builtin_amdgcn_wmma_f32_16x16x32_bf16(
            false, af[mt].v, false, bfv[nt].v, (short)0, acc[mt][nt], false, false);
  }

  #pragma unroll
  for (int mt = 0; mt < 2; ++mt)
    for (int nt = 0; nt < 4; ++nt)
      for (int i = 0; i < 8; ++i) {
        const int grow = row0 + wm * 32 + mt * 16 + i + 8 * hi;
        const int gcol = col0 + wn * 64 + nt * 16 + r;
        const size_t idx = (size_t)grow * 512 + gcol;
        float v = acc[mt][nt][i];
        if (ACC)  v += accin[idx];
        if (BIAS) v += bias[gcol];
        if (TANH) v = tanhf(v);
        if (OUTB16) outb[idx] = f2bf(v);
        else        outf[idx] = v;
      }
}

// ---------- batched aggregation GEMM: Agg_g = Ahat_g(64x64) @ H_g(64x512) ----------
constexpr uint32_t AGG_ABYTES = 64 * 144;       // Ahat tile (144B row stride)
constexpr uint32_t AGG_HPOFF  = AGG_ABYTES;     // Hp: 32 x 260 words
constexpr uint32_t AGG_SMEM   = AGG_HPOFF + 32 * 260 * 4;
__global__ void __launch_bounds__(256)
agg_gemm(const unsigned short* __restrict__ ahat, const unsigned short* __restrict__ hb,
         float* __restrict__ agg) {
  extern __shared__ char smem[];
  uint32_t* Hp = (uint32_t*)(smem + AGG_HPOFF);
  const int g = blockIdx.x, tid = threadIdx.x;
  const int wave = tid >> 5, lane = tid & 31, r = lane & 15, hi = lane >> 4;

  // Ahat tile via async DMA (direct copy, 16B aligned both sides)
  #pragma unroll
  for (int t2 = 0; t2 < 2; ++t2) {
    const int task = tid + t2 * 256;
    const int ar = task >> 3, ac = task & 7;
    async_ld_b128((uint32_t)(ar * 144 + ac * 16),
                  (uint32_t)(g * 8192 + ar * 128 + ac * 16), ahat);
  }
  const unsigned short* hg = hb + (size_t)g * TURNS * 512;

  for (int half = 0; half < 2; ++half) {
    const int cbase = half * 256;
    wait_async0();
    __syncthreads();
    #pragma unroll
    for (int t2 = 0; t2 < 4; ++t2) {        // pack H rows 2k2,2k2+1 interleaved
      const int task = tid + t2 * 256;      // 32 k2-rows x 32 col-chunks of 8
      const int k2 = task >> 5, cc = task & 31;
      const uint4 xl = *(const uint4*)(hg + (size_t)(2 * k2) * 512 + cbase + cc * 8);
      const uint4 xh = *(const uint4*)(hg + (size_t)(2 * k2 + 1) * 512 + cbase + cc * 8);
      uint32_t* dp = &Hp[k2 * 260 + cc * 8];
      dp[0] = (xl.x & 0xffffu) | (xh.x << 16);
      dp[1] = (xl.x >> 16)     | (xh.x & 0xffff0000u);
      dp[2] = (xl.y & 0xffffu) | (xh.y << 16);
      dp[3] = (xl.y >> 16)     | (xh.y & 0xffff0000u);
      dp[4] = (xl.z & 0xffffu) | (xh.z << 16);
      dp[5] = (xl.z >> 16)     | (xh.z & 0xffff0000u);
      dp[6] = (xl.w & 0xffffu) | (xh.w << 16);
      dp[7] = (xl.w >> 16)     | (xh.w & 0xffff0000u);
    }
    __syncthreads();

    Frag16 bfv[2][2];                        // wave covers 32 cols of this half
    #pragma unroll
    for (int nt = 0; nt < 2; ++nt)
      for (int ks = 0; ks < 2; ++ks)
        for (int v = 0; v < 8; ++v)
          bfv[nt][ks].u[v] = Hp[((v >> 2) * 8 + hi * 4 + (v & 3) + 16 * ks) * 260 + wave * 32 + nt * 16 + r];

    #pragma unroll
    for (int mt = 0; mt < 4; ++mt) {
      Frag16 af[2];
      #pragma unroll
      for (int ks = 0; ks < 2; ++ks) {
        const char* bp = smem + (mt * 16 + r) * 144 + hi * 16 + ks * 64;
        #pragma unroll
        for (int p = 0; p < 8; ++p)
          af[ks].u[p] = *(const uint32_t*)(bp + p * 4 + ((p >= 4) ? 16 : 0));
      }
      v8f acc[2];
      for (int nt = 0; nt < 2; ++nt)
        for (int e = 0; e < 8; ++e) acc[nt][e] = 0.0f;
      #pragma unroll
      for (int nt = 0; nt < 2; ++nt)
        for (int ks = 0; ks < 2; ++ks)
          acc[nt] = __builtin_amdgcn_wmma_f32_16x16x32_bf16(
              false, af[ks].v, false, bfv[nt][ks].v, (short)0, acc[nt], false, false);
      #pragma unroll
      for (int nt = 0; nt < 2; ++nt)
        for (int i = 0; i < 8; ++i) {
          const int lrow = mt * 16 + i + 8 * hi;
          if (lrow < TURNS)
            agg[((size_t)g * TURNS + lrow) * 512 + cbase + wave * 32 + nt * 16 + r] = acc[nt][i];
        }
    }
  }
}

// ---------- BatchNorm stats: deterministic two-stage column reduction ----------
__global__ void col_stats(const float2* __restrict__ agg2, float* __restrict__ partials) {
  const int blk = blockIdx.x, tid = threadIdx.x;     // 100 blocks x 1024 rows, 2 cols/thread
  const size_t base = (size_t)blk * 1024 * 256;
  float s0 = 0, q0 = 0, s1 = 0, q1 = 0;
  for (int rr = 0; rr < 1024; ++rr) {
    const float2 a = agg2[base + (size_t)rr * 256 + tid];
    s0 += a.x; q0 += a.x * a.x; s1 += a.y; q1 += a.y * a.y;
  }
  partials[blk * 512 + 2 * tid]     = s0;
  partials[blk * 512 + 2 * tid + 1] = s1;
  partials[100 * 512 + blk * 512 + 2 * tid]     = q0;
  partials[100 * 512 + blk * 512 + 2 * tid + 1] = q1;
}

__global__ void bn_finalize(const float* __restrict__ partials, const float* __restrict__ gamma,
                            const float* __restrict__ beta, float* __restrict__ scale,
                            float* __restrict__ shift) {
  const int c = threadIdx.x;
  float s = 0, q = 0;
  for (int b = 0; b < 100; ++b) { s += partials[b * 512 + c]; q += partials[100 * 512 + b * 512 + c]; }
  const float mean = s / (float)NNODE;
  const float var  = q / (float)NNODE - mean * mean;
  const float sc   = gamma[c] * rsqrtf(var + BNEPS);
  scale[c] = sc;
  shift[c] = beta[c] - mean * sc;
}

__global__ void bn_apply(const float4* __restrict__ agg4, const float* __restrict__ scale,
                         const float* __restrict__ shift, ushort4* __restrict__ xi4) {
  const long total = (long)NNODE * 128;
  for (long i = (long)blockIdx.x * blockDim.x + threadIdx.x; i < total;
       i += (long)gridDim.x * blockDim.x) {
    const int c = ((int)(i & 127)) << 2;
    const float4 a = agg4[i];
    ushort4 o;
    o.x = f2bf(fmaxf(scale[c + 0] * a.x + shift[c + 0], 0.0f));
    o.y = f2bf(fmaxf(scale[c + 1] * a.y + shift[c + 1], 0.0f));
    o.z = f2bf(fmaxf(scale[c + 2] * a.z + shift[c + 2], 0.0f));
    o.w = f2bf(fmaxf(scale[c + 3] * a.w + shift[c + 3], 0.0f));
    xi4[i] = o;
  }
}

// ---------- host ----------
extern "C" void kernel_launch(void* const* d_in, const int* in_sizes, int n_in,
                              void* d_out, int out_size, void* d_ws, size_t ws_size,
                              hipStream_t stream) {
  const float* x      = (const float*)d_in[0];
  const int*   ei     = (const int*)  d_in[1];
  const float* ew     = (const float*)d_in[2];
  const int*   pos    = (const int*)  d_in[3];
  const float* posemb = (const float*)d_in[4];
  const float* W1 = (const float*)d_in[5];
  const float* g1 = (const float*)d_in[7];  const float* be1 = (const float*)d_in[8];
  const float* W2 = (const float*)d_in[9];
  const float* g2 = (const float*)d_in[11]; const float* be2 = (const float*)d_in[12];
  const float* W3 = (const float*)d_in[13];
  const float* g3 = (const float*)d_in[15]; const float* be3 = (const float*)d_in[16];
  const float* Wl = (const float*)d_in[17]; const float* bl  = (const float*)d_in[18];
  const int E = in_sizes[1] / 2;
  const int* esrc = ei;
  const int* edst = ei + E;
  float* outp = (float*)d_out;

  char* ws = (char*)d_ws;
  size_t o = 0;
  auto alloc = [&](size_t bytes) { size_t p = o; o += (bytes + 255) & ~(size_t)255; return p; };
  unsigned short* xc   = (unsigned short*)(ws + alloc((size_t)NNODE * K1F * 2));      // also Xi arena
  unsigned short* hb   = (unsigned short*)(ws + alloc((size_t)(NNODE + 64) * 512 * 2));
  float*          agg  = (float*)(ws + alloc((size_t)NNODE * 512 * 4));
  unsigned short* ahat = (unsigned short*)(ws + alloc((size_t)BGRAPH * 64 * 64 * 2));
  uint32_t* wq1 = (uint32_t*)(ws + alloc((size_t)(K1F / 32) * 8192 * 4));
  uint32_t* wq2 = (uint32_t*)(ws + alloc((size_t)16 * 8192 * 4));
  uint32_t* wq3 = (uint32_t*)(ws + alloc((size_t)16 * 8192 * 4));
  uint32_t* wql = (uint32_t*)(ws + alloc((size_t)16 * 8192 * 4));
  float* partials = (float*)(ws + alloc((size_t)2 * 100 * 512 * 4));
  float* scale    = (float*)(ws + alloc(512 * 4));
  float* shift    = (float*)(ws + alloc(512 * 4));
  unsigned short* xi = xc;   // alias: xc dead after layer-1 feature GEMM

  const dim3 ggrid(4, 800);  // 512/128 col-blocks x 102400/128 row-blocks
  const uint32_t gemm_lds = 2 * ABUF;

  pack_weight<<<576, 256, 0, stream>>>(W1, wq1, K1F);
  pack_weight<<<512, 256, 0, stream>>>(W2, wq2, 512);
  pack_weight<<<512, 256, 0, stream>>>(W3, wq3, 512);
  pack_weight<<<512, 256, 0, stream>>>(Wl, wql, 512);
  build_xc<<<NNODE, 256, 0, stream>>>(x, pos, posemb, xc);
  build_adj<<<BGRAPH, 128, 0, stream>>>(esrc, edst, ew, ahat);
  zero_rows<<<64, 256, 0, stream>>>(hb + (size_t)NNODE * 512);

  auto layer = [&](const unsigned short* Ain, int K, const uint32_t* wq,
                   const float* gma, const float* bta) {
    gemm_bf16<true, false, false, false><<<ggrid, 256, gemm_lds, stream>>>(
        Ain, wq, K, hb, nullptr, nullptr, nullptr);
    agg_gemm<<<BGRAPH, 256, AGG_SMEM, stream>>>(ahat, hb, agg);
    col_stats<<<100, 256, 0, stream>>>((const float2*)agg, partials);
    bn_finalize<<<1, 512, 0, stream>>>(partials, gma, bta, scale, shift);
    bn_apply<<<4096, 256, 0, stream>>>((const float4*)agg, scale, shift, (ushort4*)xi);
  };

  // layer 1; fold x1 @ Wl (+bl) into output accumulator
  layer(xc, K1F, wq1, g1, be1);
  gemm_bf16<false, false, true, false><<<ggrid, 256, gemm_lds, stream>>>(
      xi, wql, 512, nullptr, outp, nullptr, bl);
  // layer 2; out += x2 @ Wl
  layer(xi, 512, wq2, g2, be2);
  gemm_bf16<false, true, false, false><<<ggrid, 256, gemm_lds, stream>>>(
      xi, wql, 512, nullptr, outp, outp, nullptr);
  // layer 3; out = tanh(out + x3 @ Wl)
  layer(xi, 512, wq3, g3, be3);
  gemm_bf16<false, true, false, true><<<ggrid, 256, gemm_lds, stream>>>(
      xi, wql, 512, nullptr, outp, outp, nullptr);
}
